// MultiheadAttention_22960895164556
// MI455X (gfx1250) — matrix operand order repaired and checked by reading, not compiled, not run on hardware
//
#include <hip/hip_runtime.h>
#include <hip/hip_bf16.h>

// ---------------------------------------------------------------------------
// MI455X (gfx1250) multihead attention forward, bf16 WMMA + f32 accumulate.
// B=4, S=2048, D=1024, H=16, d=64. ~137 GFLOP vs ~90MB of I/O -> compute
// bound; all GEMM-shaped work runs on v_wmma_f32_16x16x32_bf16, and GEMM
// staging uses gfx1250 async global->LDS copies (ASYNCcnt) double-buffered
// so data movement overlaps the matrix pipe.
// ---------------------------------------------------------------------------

typedef __bf16 bf16_t;
typedef __bf16 v16bf __attribute__((ext_vector_type(16)));
typedef float  v8f   __attribute__((ext_vector_type(8)));
typedef int    vint4 __attribute__((vector_size(16)));   // builtin's pointee type

#define WMMA_BF16(a, b, c) \
  __builtin_amdgcn_wmma_f32_16x16x32_bf16(false, (a), false, (b), (short)0, (c), false, false)

#ifndef HAVE_ASYNC_LDS
#if defined(__has_builtin)
#if __has_builtin(__builtin_amdgcn_global_load_async_to_lds_b128)
#define HAVE_ASYNC_LDS 1
#endif
#endif
#endif
#ifndef HAVE_ASYNC_LDS
#define HAVE_ASYNC_LDS 0
#endif

// 16-byte global -> LDS copy. Async path uses GLOBAL_LOAD_ASYNC_TO_LDS_B128
// (tracked by ASYNCcnt). Builtin takes generic int4* pointers (per compiler
// diagnostic); integer round-trip sheds the const qualifier.
__device__ inline void copy16_g2l(const void* g, void* l) {
#if HAVE_ASYNC_LDS
  __builtin_amdgcn_global_load_async_to_lds_b128(
      (vint4*)(uintptr_t)g, (vint4*)l, 0, 0);
#else
  *reinterpret_cast<uint4*>(l) = *reinterpret_cast<const uint4*>(g);
#endif
}

#if HAVE_ASYNC_LDS
#define ASYNC_WAIT_4() asm volatile("s_wait_asynccnt 0x4" ::: "memory")
#define ASYNC_WAIT_0() asm volatile("s_wait_asynccnt 0x0" ::: "memory")
#else
#define ASYNC_WAIT_4()
#define ASYNC_WAIT_0()
#endif

union Frag16 { v16bf v; unsigned u[8]; };
union Pack2  { unsigned u; bf16_t h[2]; };

// A fragment: 16x32 bf16 (lane = row; lanes 0-15 K-base 0, lanes 16-31 K-base
// 8; VGPR0-3: K=kb+2v, VGPR4-7: K=16+kb+2(v-4)). u32 loads pairwise-contiguous
// so the compiler fuses them into ds_load_b128.
__device__ inline v16bf load_a_frag_16x32(const bf16_t* p, int ld) {
  const int lane = threadIdx.x & 31;
  const int row  = lane & 15;
  const int kb   = (lane >> 4) << 3;   // 0 or 8
  Frag16 f;
#pragma unroll
  for (int i = 0; i < 8; ++i) {
    const int k = (i < 4) ? (kb + 2 * i) : (16 + kb + 2 * (i - 4));
    f.u[i] = *reinterpret_cast<const unsigned*>(p + row * ld + k);
  }
  return f.v;
}

// B fragment: 32x16 bf16 stored N-major (p[n*ld + k]) so (k,k+1) pairs are
// contiguous. Same K order as A: VGPR0-3: K=kb+2v, VGPR4-7: K=16+kb+2(v-4).
__device__ inline v16bf load_b_frag_32x16(const bf16_t* p, int ld) {
  const int lane = threadIdx.x & 31;
  const int n    = lane & 15;
  const int kb   = (lane >> 4) << 3;   // 0 or 8
  Frag16 f;
#pragma unroll
  for (int i = 0; i < 8; ++i) {
    const int k = (i < 4) ? (kb + 2 * i) : (16 + kb + 2 * (i - 4));
    f.u[i] = *reinterpret_cast<const unsigned*>(p + n * ld + k);
  }
  return f.v;
}

// ---------------------------------------------------------------------------
__global__ void cast_f32_to_bf16(const float* __restrict__ src,
                                 bf16_t* __restrict__ dst, int n) {
  int i = blockIdx.x * blockDim.x + threadIdx.x;
  const int stride = gridDim.x * blockDim.x;
  for (; i < n; i += stride) dst[i] = (bf16_t)src[i];
}

// ---------------------------------------------------------------------------
// C[M,N] = A[M,K] @ B[N,K]^T.  Block tile 128x128, BK=32, 256 threads =
// 8 waves in 4x2; each wave owns 32x64 = 2x4 WMMA tiles (8 WMMA / k-step).
// LDS double-buffered; next k-tile streams in via async copies while the
// matrix pipe chews on the current one.
// mode 0: scatter columns into head-major Q/K/V bf16 [B=4][H=16][S=2048][64]
// mode 1: write f32 row-major to Cout.
__global__ __launch_bounds__(256) void gemm_bf16_wmma(
    const bf16_t* __restrict__ A, const bf16_t* __restrict__ Bw,
    int M, int N, int K, int mode,
    float* __restrict__ Cout,
    bf16_t* __restrict__ Qb, bf16_t* __restrict__ Kb, bf16_t* __restrict__ Vb) {
  __shared__ __attribute__((aligned(16))) bf16_t As[2][128][32];
  __shared__ __attribute__((aligned(16))) bf16_t Bs[2][128][32];

  const int t   = threadIdx.x;
  const int wid = t >> 5;
  const int wy  = wid >> 1;                 // 0..3 -> 32-row band
  const int wx  = wid & 1;                  // 0..1 -> 64-col band
  const int m0  = blockIdx.y * 128, n0 = blockIdx.x * 128;

  v8f acc[2][4];
#pragma unroll
  for (int i = 0; i < 2; ++i)
#pragma unroll
    for (int j = 0; j < 4; ++j)
#pragma unroll
      for (int v = 0; v < 8; ++v) acc[i][j][v] = 0.0f;

  // Stage one 128x32 tile of A and B into LDS buffer `buf`.
  // 512 16B-chunks per matrix; 256 threads x 2 passes -> 4 async issues/thread
  // -> per-wave ASYNCcnt bump of 4 per stage.
  auto stage = [&](int buf, int k0) {
#pragma unroll
    for (int p = 0; p < 2; ++p) {
      const int chunk = t + p * 256;        // 0..511
      const int row   = chunk >> 2;         // 0..127
      const int c8    = (chunk & 3) * 8;    // bf16 column
      copy16_g2l(A + (size_t)(m0 + row) * K + k0 + c8, &As[buf][row][c8]);
      copy16_g2l(Bw + (size_t)(n0 + row) * K + k0 + c8, &Bs[buf][row][c8]);
    }
  };

  const int T = K / 32;
  stage(0, 0);
  for (int it = 0; it < T; ++it) {
    const int buf = it & 1;
    if (it + 1 < T) {
      stage(buf ^ 1, (it + 1) * 32);  // in flight during this iteration's WMMAs
      ASYNC_WAIT_4();                 // previous stage (4 older issues) landed
    } else {
      ASYNC_WAIT_0();
    }
    __syncthreads();

    v16bf af[2], bfr[4];
#pragma unroll
    for (int i = 0; i < 2; ++i)
      af[i] = load_a_frag_16x32(&As[buf][wy * 32 + i * 16][0], 32);
#pragma unroll
    for (int j = 0; j < 4; ++j)
      bfr[j] = load_b_frag_32x16(&Bs[buf][wx * 64 + j * 16][0], 32);
#pragma unroll
    for (int i = 0; i < 2; ++i)
#pragma unroll
      for (int j = 0; j < 4; ++j)
        acc[i][j] = WMMA_BF16(af[i], bfr[j], acc[i][j]);
    __syncthreads();                  // readers done before buf is re-staged
  }

  // Epilogue: C/D layout — lane&15 = column, VGPR index (+8 upper half) = row.
  const int lane = t & 31;
  const int nloc = lane & 15;
  const int rsel = (lane >> 4) * 8;
#pragma unroll
  for (int i = 0; i < 2; ++i)
#pragma unroll
    for (int j = 0; j < 4; ++j)
#pragma unroll
      for (int v = 0; v < 8; ++v) {
        const int m = m0 + wy * 32 + i * 16 + v + rsel;
        const int n = n0 + wx * 64 + j * 16 + nloc;
        const float val = acc[i][j][v];
        if (mode == 1) {
          Cout[(size_t)m * N + n] = val;
        } else {
          // qkv column e -> head h = e/192, j = e%192: [0,64)=Q [64,128)=K [128,192)=V
          const int b  = m >> 11;        // S = 2048
          const int s  = m & 2047;
          const int h  = n / 192;
          const int jj = n % 192;
          bf16_t* dst  = (jj < 64) ? Qb : (jj < 128 ? Kb : Vb);
          const int d  = jj & 63;
          dst[(((size_t)b * 16 + h) * 2048 + s) * 64 + d] = (bf16_t)val;
        }
      }
}

// ---------------------------------------------------------------------------
// Fused flash attention: one block = one (b,h) x 64 queries (16 per wave).
// Streams keys in blocks of 32; online softmax in f32; QK^T and P@V on WMMA.
__global__ __launch_bounds__(128) void attn_fwd(
    const bf16_t* __restrict__ Qb, const bf16_t* __restrict__ Kb,
    const bf16_t* __restrict__ Vb, bf16_t* __restrict__ vals) {
  constexpr int S = 2048, Dh = 64;
  __shared__ __attribute__((aligned(16))) bf16_t Ks[32][64];      // keys row-major
  __shared__ __attribute__((aligned(16))) bf16_t Vt[64][32];      // V transposed
  __shared__ __attribute__((aligned(16))) bf16_t Pl[4][16][32];   // per-wave P scratch

  const int t    = threadIdx.x;
  const int wid  = t >> 5;
  const int lane = t & 31;
  const int bh   = blockIdx.x;                 // b*16 + h
  const int q0   = blockIdx.y * 64 + wid * 16;

  const bf16_t* Qbase = Qb + ((size_t)bh * S + q0) * Dh;
  v16bf aq[2];
  aq[0] = load_a_frag_16x32(Qbase, Dh);        // d 0..31
  aq[1] = load_a_frag_16x32(Qbase + 32, Dh);   // d 32..63

  v8f o[4];
  float m_run[8], l_run[8];
#pragma unroll
  for (int c = 0; c < 4; ++c)
#pragma unroll
    for (int v = 0; v < 8; ++v) o[c][v] = 0.0f;
#pragma unroll
  for (int v = 0; v < 8; ++v) { m_run[v] = -__builtin_inff(); l_run[v] = 0.0f; }

  const int row_off = (lane >> 4) * 8;

  for (int kb0 = 0; kb0 < S; kb0 += 32) {
    // Stage K block 32x64 row-major (B-frag pairs over d are contiguous).
#pragma unroll
    for (int p = 0; p < 2; ++p) {
      const int chunk = t + p * 128;           // 0..255 sixteen-byte chunks
      const int row = chunk >> 3;
      const int c8  = (chunk & 7) * 8;
      *reinterpret_cast<uint4*>(&Ks[row][c8]) =
          *reinterpret_cast<const uint4*>(Kb + ((size_t)bh * S + kb0 + row) * Dh + c8);
    }
    // Stage V block transposed: Vt[dim][key] (B-frag pairs over keys contiguous).
#pragma unroll
    for (int i = 0; i < 8; ++i) {
      const int idx = t * 8 + i;               // 1024 dword pairs = 32x64 tile
      const int key = idx >> 5;
      const int dp  = (idx & 31) * 2;
      Pack2 cv;
      cv.u = *reinterpret_cast<const unsigned*>(Vb + ((size_t)bh * S + kb0 + key) * Dh + dp);
      Vt[dp][key]     = cv.h[0];
      Vt[dp + 1][key] = cv.h[1];
    }
    if (kb0 + 32 < S) {  // warm next K rows (global_prefetch_b8)
      __builtin_prefetch(Kb + ((size_t)bh * S + kb0 + 32 + (t >> 2)) * Dh, 0, 0);
      __builtin_prefetch(Vb + ((size_t)bh * S + kb0 + 32 + (t >> 2)) * Dh, 0, 0);
    }
    __syncthreads();

    // logits for two 16-key tiles: S = Q(16x64) @ K^T
    v8f s0, s1;
#pragma unroll
    for (int v = 0; v < 8; ++v) { s0[v] = 0.0f; s1[v] = 0.0f; }
    v16bf bk;
    bk = load_b_frag_32x16(&Ks[0][0],   64); s0 = WMMA_BF16(aq[0], bk, s0);
    bk = load_b_frag_32x16(&Ks[0][32],  64); s0 = WMMA_BF16(aq[1], bk, s0);
    bk = load_b_frag_32x16(&Ks[16][0],  64); s1 = WMMA_BF16(aq[0], bk, s1);
    bk = load_b_frag_32x16(&Ks[16][32], 64); s1 = WMMA_BF16(aq[1], bk, s1);

    // Online softmax: row reductions across the 16-lane halves.
    float scale[8];
#pragma unroll
    for (int v = 0; v < 8; ++v) {
      float mx = fmaxf(s0[v], s1[v]);
#pragma unroll
      for (int off = 8; off; off >>= 1) mx = fmaxf(mx, __shfl_xor(mx, off, 16));
      const float mnew = fmaxf(m_run[v], mx);
      scale[v] = __expf(m_run[v] - mnew);
      m_run[v] = mnew;
    }
#pragma unroll
    for (int v = 0; v < 8; ++v) {
      const float p0 = __expf(s0[v] - m_run[v]);
      const float p1 = __expf(s1[v] - m_run[v]);
      float rs = p0 + p1;
#pragma unroll
      for (int off = 8; off; off >>= 1) rs += __shfl_xor(rs, off, 16);
      l_run[v] = l_run[v] * scale[v] + rs;
      const int row = v + row_off;
      Pl[wid][row][lane & 15]        = (bf16_t)p0;   // key tile 0
      Pl[wid][row][16 + (lane & 15)] = (bf16_t)p1;   // key tile 1
#pragma unroll
      for (int c = 0; c < 4; ++c) o[c][v] *= scale[v];
    }

    // O += P(16x32) @ V(32x64); per-wave P scratch, same-wave DS ops in-order.
    const v16bf ap = load_a_frag_16x32(&Pl[wid][0][0], 32);
#pragma unroll
    for (int c = 0; c < 4; ++c) {
      const v16bf bv = load_b_frag_32x16(&Vt[c * 16][0], 32);
      o[c] = WMMA_BF16(ap, bv, o[c]);
    }
    __syncthreads();
  }

  // Normalize and write vals[b][s][h*64 + d] (bf16, feeds output projection).
  const int b = bh >> 4, h = bh & 15;
  const int nloc = lane & 15;
#pragma unroll
  for (int c = 0; c < 4; ++c)
#pragma unroll
    for (int v = 0; v < 8; ++v) {
      const int row = v + row_off;
      const float ov = o[c][v] / l_run[v];
      const int s = q0 + row;
      vals[((size_t)b * 2048 + s) * 1024 + h * 64 + c * 16 + nloc] = (bf16_t)ov;
    }
}

// ---------------------------------------------------------------------------
extern "C" void kernel_launch(void* const* d_in, const int* in_sizes, int n_in,
                              void* d_out, int out_size, void* d_ws, size_t ws_size,
                              hipStream_t stream) {
  (void)in_sizes; (void)n_in; (void)out_size; (void)ws_size;
  const float* x     = (const float*)d_in[0];   // [4,2048,1024]
  const float* w_qkv = (const float*)d_in[1];   // [3072,1024]
  const float* w_out = (const float*)d_in[2];   // [1024,1024]
  float* out = (float*)d_out;                   // [4,2048,1024]

  constexpr size_t MiB = 1u << 20;
  char* ws = (char*)d_ws;
  bf16_t* xb    = (bf16_t*)(ws + 0);            // 16 MiB
  bf16_t* wqkvb = (bf16_t*)(ws + 16 * MiB);     //  6 MiB
  bf16_t* woutb = (bf16_t*)(ws + 22 * MiB);     //  2 MiB
  bf16_t* Qb    = (bf16_t*)(ws + 24 * MiB);     // 16 MiB  [B,H,S,64]
  bf16_t* Kb    = (bf16_t*)(ws + 40 * MiB);     // 16 MiB
  bf16_t* Vb    = (bf16_t*)(ws + 56 * MiB);     // 16 MiB
  bf16_t* valsb = (bf16_t*)(ws + 72 * MiB);     // 16 MiB  [B,S,D]

  cast_f32_to_bf16<<<4096, 256, 0, stream>>>(x,     xb,    4 * 2048 * 1024);
  cast_f32_to_bf16<<<2048, 256, 0, stream>>>(w_qkv, wqkvb, 3072 * 1024);
  cast_f32_to_bf16<<<1024, 256, 0, stream>>>(w_out, woutb, 1024 * 1024);

  // qkv = x @ w_qkv^T, scatter to head-major Q/K/V
  dim3 g1(3072 / 128, 8192 / 128);
  gemm_bf16_wmma<<<g1, 256, 0, stream>>>(xb, wqkvb, 8192, 3072, 1024, 0,
                                         nullptr, Qb, Kb, Vb);
  // fused flash attention
  dim3 g2(64, 2048 / 64);
  attn_fwd<<<g2, 128, 0, stream>>>(Qb, Kb, Vb, valsb);
  // out = vals @ w_out^T (f32)
  dim3 g3(1024 / 128, 8192 / 128);
  gemm_bf16_wmma<<<g3, 256, 0, stream>>>(valsb, woutb, 8192, 1024, 1024, 1,
                                         out, nullptr, nullptr, nullptr);
}
